// SelfAttentionPooling_58574763983146
// MI455X (gfx1250) — matrix operand order repaired and checked by reading, not compiled, not run on hardware
//
#include <hip/hip_runtime.h>

typedef __attribute__((ext_vector_type(16))) _Float16 v16h;
typedef __attribute__((ext_vector_type(8)))  _Float16 v8h;
typedef __attribute__((ext_vector_type(8)))  float    v8f;

union F16Frag { v16h v; v8h h[2]; };

#define WMMA_F16(a, b, c) \
  __builtin_amdgcn_wmma_f32_16x16x32_f16(false, (a), false, (b), (short)0, (c), false, false)

#define G  4            // batches per workgroup
#define HS (16 * 512)   // elements in one 16x512 f16 matrix (== 512x16)

__device__ inline v8f zero8f() {
  v8f z;
#pragma unroll
  for (int q = 0; q < 8; ++q) z[q] = 0.f;
  return z;
}

// Convert the three 512x512 f32 weight matrices to f16, concatenated [Q|K|V].
__global__ void wcvt_kernel(const float* __restrict__ wq, const float* __restrict__ wk,
                            const float* __restrict__ wv, _Float16* __restrict__ o) {
  const int N = 512 * 512;
  int i = blockIdx.x * blockDim.x + threadIdx.x;
  if (i >= 3 * N) return;
  float x;
  if (i < N)          x = wq[i];
  else if (i < 2 * N) x = wk[i - N];
  else                x = wv[i - 2 * N];
  o[i] = (_Float16)x;
}

__launch_bounds__(1024)
__global__ void attnpool_kernel(const float* __restrict__ H, const float* __restrict__ F,
                                const _Float16* __restrict__ Wh, float* __restrict__ out) {
  // Dynamic LDS (~268.5 KB, fits gfx1250's 320 KB WGP LDS):
  //   Hs   [G][16][512] f16
  //   Qs   [G][16][512] f16
  //   Ks   [G][16][512] f16
  //   Vt   [G][512][16] f16  -- V transposed
  //   Al   [G][16][32]  f16  -- alpha, cols 16..31 zero-padded
  //   hpool[G][512]     f32  -- sum_n F[n]*H[n][e] (residual's pooled term)
  //   Fl, LFl [G][16]   f32
  extern __shared__ __align__(16) char smem_raw[];
  _Float16* Hs = (_Float16*)smem_raw;
  _Float16* Qs = Hs + G * HS;
  _Float16* Ks = Qs + G * HS;
  _Float16* Vt = Ks + G * HS;
  _Float16* Al = Vt + G * HS;
  float* hpool = (float*)(Al + G * 16 * 32);
  float* Fl    = hpool + G * 512;
  float* LFl   = Fl + G * 16;

  const int t     = threadIdx.x;
  const int lane  = t & 31;
  const int w     = t >> 5;       // wave id 0..31
  const int row   = lane & 15;    // A-row / B-col / D-col within tile
  const int khalf = lane >> 4;
  const int kbase = khalf * 16;   // B-matrix K base for this lane
  const int b0    = blockIdx.x * G;

  if (t < 16 * G) {
    int g = t >> 4, n = t & 15;
    float f = F[(b0 + g) * 16 + n];
    Fl[t]  = f;
    LFl[t] = __logf(f + 1e-8f);
  }
  __syncthreads();

  // ---- Stage H -> LDS f16, accumulate residual's pooled term in f32 ----
  {
    const int e  = t & 511;
    const int gh = t >> 9;        // 0..1; thread covers batches {2gh, 2gh+1}
#pragma unroll
    for (int gg = 0; gg < 2; ++gg) {
      const int g = gh * 2 + gg;
      const float* Hb = H + (size_t)(b0 + g) * HS;
      float hp = 0.f;
#pragma unroll
      for (int n = 0; n < 16; ++n) {
        float h = __builtin_nontemporal_load(&Hb[n * 512 + e]);
        Hs[g * HS + n * 512 + e] = (_Float16)h;
        hp += Fl[g * 16 + n] * h;
      }
      hpool[g * 512 + e] = hp;
    }
  }
  __syncthreads();

  // ---- QKV GEMM in 3 passes (pass = 0:Q, 1:K, 2:V); each wave owns e-tile w ----
  // Per (pass,kk): one weight B-frag feeds G WMMAs; all G A-frags loaded into
  // distinct registers first so the WMMAs overlap the LDS loads (partial waits).
  const int e0 = w * 16;
#pragma unroll
  for (int pass = 0; pass < 3; ++pass) {
    const _Float16* wbase = Wh + pass * (512 * 512) + (e0 + row) * 512;
    v8f acc[G];
#pragma unroll
    for (int g = 0; g < G; ++g) acc[g] = zero8f();

    for (int kk = 0; kk < 16; ++kk) {
      const int k0 = kk * 32;
      F16Frag bf;
      bf.h[0] = *(const v8h*)(wbase + k0 + kbase);
      bf.h[1] = *(const v8h*)(wbase + k0 + kbase + 8);
      F16Frag a[G];
#pragma unroll
      for (int g = 0; g < G; ++g) {
        a[g].h[0] = *(const v8h*)&Hs[g * HS + row * 512 + k0 + kbase];
        a[g].h[1] = *(const v8h*)&Hs[g * HS + row * 512 + k0 + kbase + 8];
      }
#pragma unroll
      for (int g = 0; g < G; ++g)
        acc[g] = WMMA_F16(a[g].v, bf.v, acc[g]);
    }

    // Writeout (uniform per pass): Q,K row-major; V transposed packed b128.
#pragma unroll
    for (int g = 0; g < G; ++g) {
      if (pass == 2) {
        v8h pk;
#pragma unroll
        for (int r = 0; r < 8; ++r) pk[r] = (_Float16)acc[g][r];
        *(v8h*)&Vt[g * HS + (e0 + row) * 16 + khalf * 8] = pk;   // V^T[e][m]
      } else {
        _Float16* dst = ((pass == 0) ? Qs : Ks) + g * HS;
#pragma unroll
        for (int r = 0; r < 8; ++r)
          dst[(khalf * 8 + r) * 512 + e0 + row] = (_Float16)acc[g][r];
      }
    }
  }
  __syncthreads();

  // ---- Scores + softmax: wave g handles batch g ----
  if (w < G) {
    const int g = w;
    v8f sc = zero8f();
    for (int kk = 0; kk < 16; ++kk) {
      const int k0 = kk * 32;
      F16Frag a, bf;
      a.h[0]  = *(const v8h*)&Qs[g * HS + row * 512 + k0 + kbase];
      a.h[1]  = *(const v8h*)&Qs[g * HS + row * 512 + k0 + kbase + 8];
      bf.h[0] = *(const v8h*)&Ks[g * HS + row * 512 + k0 + kbase];
      bf.h[1] = *(const v8h*)&Ks[g * HS + row * 512 + k0 + kbase + 8];
      sc = WMMA_F16(a.v, bf.v, sc);
    }
    const float scale = 0.04419417382415922f;  // 1/sqrt(512)
    const float lf = LFl[g * 16 + row];        // key index == D-tile column
#pragma unroll
    for (int r = 0; r < 8; ++r) {
      float s = sc[r] * scale + lf;
      float mx = s;
      mx = fmaxf(mx, __shfl_xor(mx, 1, 32));
      mx = fmaxf(mx, __shfl_xor(mx, 2, 32));
      mx = fmaxf(mx, __shfl_xor(mx, 4, 32));
      mx = fmaxf(mx, __shfl_xor(mx, 8, 32));
      float e = __expf(s - mx);
      float sm = e;
      sm += __shfl_xor(sm, 1, 32);
      sm += __shfl_xor(sm, 2, 32);
      sm += __shfl_xor(sm, 4, 32);
      sm += __shfl_xor(sm, 8, 32);
      float al = e / sm;
      int m = khalf * 8 + r;                   // query row
      Al[g * 512 + m * 32 + row]      = (_Float16)al;
      Al[g * 512 + m * 32 + 16 + row] = (_Float16)0.f;
    }
  }
  __syncthreads();

  // ---- h_attn = alpha @ V (4 feature tiles per wave), pool, residual, store ----
  {
    const int g = w >> 3;                      // waves 0-7 -> g0, ... 24-31 -> g3
    F16Frag af;
    af.h[0] = *(const v8h*)&Al[g * 512 + row * 32 + kbase];
    af.h[1] = *(const v8h*)&Al[g * 512 + row * 32 + kbase + 8];
    float fw[8];
#pragma unroll
    for (int r = 0; r < 8; ++r) fw[r] = Fl[g * 16 + khalf * 8 + r];

#pragma unroll
    for (int j = 0; j < 4; ++j) {
      const int f0 = ((w & 7) * 4 + j) * 16;   // feature tile base
      F16Frag bf;
      if (lane < 16) {  // real keys 0..15; lanes>=16 carry the zero pad (K 16..31)
        bf.h[0] = *(const v8h*)&Vt[g * HS + (f0 + row) * 16];
        bf.h[1] = *(const v8h*)&Vt[g * HS + (f0 + row) * 16 + 8];
      } else {
#pragma unroll
        for (int q = 0; q < 8; ++q) { bf.h[0][q] = (_Float16)0.f; bf.h[1][q] = (_Float16)0.f; }
      }
      v8f ha = zero8f();
      ha = WMMA_F16(af.v, bf.v, ha);
      float p = 0.f;
#pragma unroll
      for (int r = 0; r < 8; ++r) p += fw[r] * ha[r];   // this lane's 8 query rows
      p += __shfl_xor(p, 16, 32);                        // other 8 rows, same column
      if (lane < 16) {
        out[(size_t)(b0 + g) * 512 + f0 + lane] = p + hpool[g * 512 + f0 + lane];
      }
    }
  }
}

extern "C" void kernel_launch(void* const* d_in, const int* in_sizes, int n_in,
                              void* d_out, int out_size, void* d_ws, size_t ws_size,
                              hipStream_t stream) {
  const float* H  = (const float*)d_in[0];
  const float* F  = (const float*)d_in[1];
  const float* WQ = (const float*)d_in[2];
  const float* WK = (const float*)d_in[3];
  const float* WV = (const float*)d_in[4];
  float* out      = (float*)d_out;
  _Float16* Wh    = (_Float16*)d_ws;    // needs 3*512*512*2 = 1.5 MB

  const int B = in_sizes[1] / 16;       // F is [B,16]

  const size_t smem = (size_t)G * HS * 2 * 4        // Hs, Qs, Ks, Vt (f16)
                    + (size_t)G * 16 * 32 * 2       // alpha
                    + (size_t)G * 512 * 4           // hpool
                    + (size_t)G * 16 * 4 * 2;       // Fl, LFl

  wcvt_kernel<<<(3 * 512 * 512 + 255) / 256, 256, 0, stream>>>(WQ, WK, WV, Wh);
  attnpool_kernel<<<B / G, 1024, smem, stream>>>(H, F, Wh, out);
}
